// MultiHeadAttention_49572512530888
// MI455X (gfx1250) — matrix operand order repaired and checked by reading, not compiled, not run on hardware
//
#include <hip/hip_runtime.h>

// ---------------------------------------------------------------------------
// MHA forward for MI455X (gfx1250, wave32, WMMA).
//   B=2, S=2048, D_MODEL=1024, H=16, d_h=64
// Pipeline:
//   1) cvt: f32 -> f16 for Q,K,V and the 4 weight matrices (into d_ws)
//   2) gemm_proj<0..2>: q/k/v = X @ W^T + b  (f16 WMMA, K-loop over 1024)
//        q scaled by 0.125*log2(e)  (softmax runs in exp2 domain);
//        q,k stored [B,H,S,64]; v stored transposed [B,H,64,S]
//   3) flash_attn: per-wave 16-query flash attention over 64-key blocks
//        (QK^T WMMA -> online exp2-softmax via shfl_xor -> LDS relayout
//         -> PV WMMA), with global_prefetch of the next key/value block
//   4) gemm_proj<3>: out = ctx @ Wo^T + b  (f32 output to d_out)
// ---------------------------------------------------------------------------

typedef __attribute__((ext_vector_type(16))) _Float16 v16h;
typedef __attribute__((ext_vector_type(8)))  _Float16 v8h;
typedef __attribute__((ext_vector_type(4)))  _Float16 v4h;
typedef __attribute__((ext_vector_type(8)))  float    v8f;

#define D_MODEL 1024
#define N_HEADS 16
#define D_H     64
#define SEQ     2048
#define MTOT    4096        // B * S

// 1/sqrt(d_h) * log2(e): scores land directly in the exp2 domain.
#define QSCALE  0.18033688011112042f

__device__ __forceinline__ v8f wmma_f16(v16h a, v16h b, v8f c) {
  return __builtin_amdgcn_wmma_f32_16x16x32_f16(false, a, false, b,
                                                (short)0, c, false, false);
}

// A-fragment (16x32 f16): lane holds row = lane%16; element k-packing is
// {16*half+0..15} (contiguous, matches B-frags) -> two 16B loads.
__device__ __forceinline__ v16h load_afrag(const _Float16* p, int kb8) {
  v16h r;
  *(v8h*)&r       = *(const v8h*)(p + 2 * kb8);
  *((v8h*)&r + 1) = *(const v8h*)(p + 2 * kb8 + 8);
  return r;
}

// ---------------------------------------------------------------------------
// 1) f32 -> f16 conversion (vectorized x4)
// ---------------------------------------------------------------------------
__global__ __launch_bounds__(256) void cvt_f32_f16(
    const float* __restrict__ src, _Float16* __restrict__ dst, int n) {
  int i = (blockIdx.x * 256 + threadIdx.x) * 4;
  if (i < n) {
    const float4 f = *(const float4*)(src + i);
    v4h h;
    h[0] = (_Float16)f.x; h[1] = (_Float16)f.y;
    h[2] = (_Float16)f.z; h[3] = (_Float16)f.w;
    *(v4h*)(dst + i) = h;
  }
}

// ---------------------------------------------------------------------------
// 2/4) GEMM  Y = X @ W^T + bias.  One wave computes a 16(M) x 64(N) strip,
// K-loop of 32 (one A-frag reused across 4 WMMAs).  MODE selects epilogue:
//   0: q  -> f16 [B,H,S,64], scaled by QSCALE
//   1: k  -> f16 [B,H,S,64]
//   2: v  -> f16 [B,H,64,S]  (transposed for the PV B-fragment)
//   3: f32 plain [M, D_MODEL] (final output)
// ---------------------------------------------------------------------------
template <int MODE>
__global__ __launch_bounds__(256) void gemm_proj(
    const _Float16* __restrict__ X, const _Float16* __restrict__ W,
    const float* __restrict__ bias, void* __restrict__ outp) {
  const int lane = threadIdx.x & 31;
  const int wv   = threadIdx.x >> 5;
  const int wg   = blockIdx.x * 8 + wv;     // global wave id: 0..4095
  const int mt   = wg >> 4;                 // M tile (16 rows), 0..255
  const int nb   = wg & 15;                 // 64-wide N strip (== head)
  const int col  = lane & 15;
  const int lh   = lane >> 4;               // lane-group half
  const int kb8  = lh * 8;

  const _Float16* arow = X + (size_t)(mt * 16 + col) * D_MODEL;
  v8f acc[4] = {};
  for (int k0 = 0; k0 < D_MODEL; k0 += 32) {
    if (k0 + 64 < D_MODEL)
      __builtin_prefetch(arow + k0 + 64, 0, 1);   // global_prefetch_b8
    const v16h a = load_afrag(arow + k0, kb8);
#pragma unroll
    for (int j = 0; j < 4; ++j) {
      // B-frag: lane = column n = W row; K = 16*lh + 0..15 contiguous.
      const _Float16* bp =
          W + (size_t)(nb * 64 + j * 16 + col) * D_MODEL + k0 + 16 * lh;
      const v16h b = *(const v16h*)bp;
      acc[j] = wmma_f16(a, b, acc[j]);
    }
  }
  // Epilogue: element (m, n): m = mt*16 + r + 8*lh, n = nb*64 + j*16 + col.
#pragma unroll
  for (int j = 0; j < 4; ++j) {
    const float bv = bias[nb * 64 + j * 16 + col];
#pragma unroll
    for (int r = 0; r < 8; ++r) {
      const int m = mt * 16 + r + 8 * lh;
      float v = acc[j][r] + bv;
      if (MODE == 0) v *= QSCALE;          // 1/sqrt(d_h)*log2e folded into q
      const int dh = j * 16 + col;
      if (MODE <= 1) {
        const int b = m >> 11, s = m & (SEQ - 1);
        ((_Float16*)outp)[((size_t)(b * N_HEADS + nb) * SEQ + s) * D_H + dh] =
            (_Float16)v;
      } else if (MODE == 2) {
        const int b = m >> 11, s = m & (SEQ - 1);
        ((_Float16*)outp)[((size_t)(b * N_HEADS + nb) * D_H + dh) * SEQ + s] =
            (_Float16)v;
      } else {
        ((float*)outp)[(size_t)m * D_MODEL + nb * 64 + dh] = v;
      }
    }
  }
}

// ---------------------------------------------------------------------------
// 3) Flash attention.  grid = (16, B*H), block = 256 (8 waves).
// Each wave owns 16 query rows; streams causal key blocks of 64
// (4 score tiles -> one softmax update -> 8 PV WMMAs per block).
// ---------------------------------------------------------------------------
__global__ __launch_bounds__(256) void flash_attn(
    const _Float16* __restrict__ qh, const _Float16* __restrict__ kh,
    const _Float16* __restrict__ vTh, _Float16* __restrict__ ctxh) {
  __shared__ _Float16 lds[8 * 16 * 64];     // 2KB P-staging per wave
  const int lane = threadIdx.x & 31;
  const int wv   = threadIdx.x >> 5;
  const int bh   = blockIdx.y;
  const int qb   = (blockIdx.x * 8 + wv) * 16;
  const int col  = lane & 15;
  const int lh   = lane >> 4;
  const int kb8  = lh * 8;
  _Float16* sw = lds + wv * (16 * 64);

  // q A-fragments for dh 0..31 and 32..63 (held for the whole key loop)
  const _Float16* qrow = qh + ((size_t)bh * SEQ + qb + col) * D_H;
  const v16h qa0 = load_afrag(qrow, kb8);
  const v16h qa1 = load_afrag(qrow + 32, kb8);

  v8f ctx[4] = {};
  float mrun[8], lrun[8];
#pragma unroll
  for (int r = 0; r < 8; ++r) { mrun[r] = -3.0e38f; lrun[r] = 0.0f; }

  const int kend = ((qb + 79) >> 6) << 6;   // causal: keys 0 .. qb+15
  for (int kb = 0; kb < kend; kb += 64) {
    const bool more = (kb + 64) < kend;
    // ---- scores: four 16x16 tiles (keys kb+16t .. kb+16t+15), K=64 ----
    v8f s[4];
    const _Float16* kbase = kh + ((size_t)bh * SEQ + kb + col) * D_H + 16 * lh;
#pragma unroll
    for (int t = 0; t < 4; ++t) {
      const _Float16* kp = kbase + (size_t)t * 16 * D_H;
      v8f a = {};
      a = wmma_f16(qa0, *(const v16h*)kp, a);
      a = wmma_f16(qa1, *(const v16h*)(kp + 32), a);
      s[t] = a;
    }
    if (more)  // pull next key block toward the WGP while softmax runs
      __builtin_prefetch(kbase + (size_t)64 * D_H, 0, 1);
    // ---- causal mask on diagonal blocks ----
    if (kb + 63 > qb) {
#pragma unroll
      for (int t = 0; t < 4; ++t)
#pragma unroll
        for (int r = 0; r < 8; ++r)
          if (kb + 16 * t + col > qb + r + 8 * lh) s[t][r] = -1.0e30f;
    }
    // ---- online exp2-softmax (row = 16 lanes inside each half-wave) ----
    float fscale[8];
#pragma unroll
    for (int r = 0; r < 8; ++r) {
      float t = fmaxf(fmaxf(s[0][r], s[1][r]), fmaxf(s[2][r], s[3][r]));
      t = fmaxf(t, __shfl_xor(t, 1));
      t = fmaxf(t, __shfl_xor(t, 2));
      t = fmaxf(t, __shfl_xor(t, 4));
      t = fmaxf(t, __shfl_xor(t, 8));
      const float mnew = fmaxf(mrun[r], t);
      fscale[r] = __builtin_amdgcn_exp2f(mrun[r] - mnew);
      mrun[r] = mnew;
      const float p0 = __builtin_amdgcn_exp2f(s[0][r] - mnew);
      const float p1 = __builtin_amdgcn_exp2f(s[1][r] - mnew);
      const float p2 = __builtin_amdgcn_exp2f(s[2][r] - mnew);
      const float p3 = __builtin_amdgcn_exp2f(s[3][r] - mnew);
      float rs = (p0 + p1) + (p2 + p3);
      rs += __shfl_xor(rs, 1);
      rs += __shfl_xor(rs, 2);
      rs += __shfl_xor(rs, 4);
      rs += __shfl_xor(rs, 8);
      lrun[r] = lrun[r] * fscale[r] + rs;
      // stage P (D-layout -> LDS row-major 16x64)
      _Float16* swr = sw + (r + 8 * lh) * 64 + col;
      swr[0]  = (_Float16)p0;
      swr[16] = (_Float16)p1;
      swr[32] = (_Float16)p2;
      swr[48] = (_Float16)p3;
    }
#pragma unroll
    for (int j = 0; j < 4; ++j)
#pragma unroll
      for (int r = 0; r < 8; ++r) ctx[j][r] *= fscale[r];

    // reload P as A-fragments (same-wave LDS, DScnt waits by compiler)
    const _Float16* prow = sw + col * 64;
    const v16h pa0 = load_afrag(prow, kb8);
    const v16h pa1 = load_afrag(prow + 32, kb8);
    // ---- ctx += P @ V[kb..kb+63, :]  (vT gives contiguous B-frags) ----
#pragma unroll
    for (int j = 0; j < 4; ++j) {
      const _Float16* vp =
          vTh + ((size_t)bh * D_H + j * 16 + col) * SEQ + kb + 16 * lh;
      if (more) __builtin_prefetch(vp + 64, 0, 1);
      ctx[j] = wmma_f16(pa0, *(const v16h*)vp, ctx[j]);
      ctx[j] = wmma_f16(pa1, *(const v16h*)(vp + 32), ctx[j]);
    }
  }
  // ---- normalize and store merged-head ctx (f16 [B,S,1024]) ----
  float linv[8];
#pragma unroll
  for (int r = 0; r < 8; ++r) linv[r] = 1.0f / lrun[r];
  const int b = bh >> 4, h = bh & 15;
#pragma unroll
  for (int j = 0; j < 4; ++j) {
#pragma unroll
    for (int r = 0; r < 8; ++r) {
      const int m = qb + r + 8 * lh;
      ctxh[(size_t)(b * SEQ + m) * D_MODEL + h * 64 + j * 16 + col] =
          (_Float16)(ctx[j][r] * linv[r]);
    }
  }
}

// ---------------------------------------------------------------------------
extern "C" void kernel_launch(void* const* d_in, const int* in_sizes, int n_in,
                              void* d_out, int out_size, void* d_ws,
                              size_t ws_size, hipStream_t stream) {
  (void)in_sizes; (void)n_in; (void)out_size; (void)ws_size;
  const float* Q   = (const float*)d_in[0];
  const float* K   = (const float*)d_in[1];
  const float* V   = (const float*)d_in[2];
  // d_in[3] = causal mask; recomputed analytically in flash_attn
  const float* WQw = (const float*)d_in[4];
  const float* WQb = (const float*)d_in[5];
  const float* WKw = (const float*)d_in[6];
  const float* WKb = (const float*)d_in[7];
  const float* WVw = (const float*)d_in[8];
  const float* WVb = (const float*)d_in[9];
  const float* Wow = (const float*)d_in[10];
  const float* Wob = (const float*)d_in[11];

  const size_t SZ_X = (size_t)MTOT * D_MODEL;      // 4,194,304 halves
  const size_t SZ_W = (size_t)D_MODEL * D_MODEL;   // 1,048,576 halves
  _Float16* Qh   = (_Float16*)d_ws;
  _Float16* Kh   = Qh + SZ_X;
  _Float16* Vh   = Kh + SZ_X;
  _Float16* WQh  = Vh + SZ_X;
  _Float16* WKh  = WQh + SZ_W;
  _Float16* WVh  = WKh + SZ_W;
  _Float16* Woh  = WVh + SZ_W;
  _Float16* qhd  = Woh + SZ_W;   // [B,H,S,64]   (pre-scaled by QSCALE)
  _Float16* khd  = qhd + SZ_X;   // [B,H,S,64]
  _Float16* vTd  = khd + SZ_X;   // [B,H,64,S]
  _Float16* ctxh = vTd + SZ_X;   // [B,S,1024]

  const dim3 blk(256);
  cvt_f32_f16<<<(int)(SZ_X / 1024), blk, 0, stream>>>(Q, Qh, (int)SZ_X);
  cvt_f32_f16<<<(int)(SZ_X / 1024), blk, 0, stream>>>(K, Kh, (int)SZ_X);
  cvt_f32_f16<<<(int)(SZ_X / 1024), blk, 0, stream>>>(V, Vh, (int)SZ_X);
  cvt_f32_f16<<<(int)(SZ_W / 1024), blk, 0, stream>>>(WQw, WQh, (int)SZ_W);
  cvt_f32_f16<<<(int)(SZ_W / 1024), blk, 0, stream>>>(WKw, WKh, (int)SZ_W);
  cvt_f32_f16<<<(int)(SZ_W / 1024), blk, 0, stream>>>(WVw, WVh, (int)SZ_W);
  cvt_f32_f16<<<(int)(SZ_W / 1024), blk, 0, stream>>>(Wow, Woh, (int)SZ_W);

  gemm_proj<0><<<512, blk, 0, stream>>>(Qh, WQh, WQb, qhd);
  gemm_proj<1><<<512, blk, 0, stream>>>(Kh, WKh, WKb, khd);
  gemm_proj<2><<<512, blk, 0, stream>>>(Vh, WVh, WVb, vTd);

  flash_attn<<<dim3(16, 32), blk, 0, stream>>>(qhd, khd, vTd, ctxh);

  gemm_proj<3><<<512, blk, 0, stream>>>(ctxh, Woh, Wob, d_out);
}
